// MultiHeadAttention_23974507446726
// MI455X (gfx1250) — matrix operand order repaired and checked by reading, not compiled, not run on hardware
//
#include <hip/hip_runtime.h>

typedef __attribute__((ext_vector_type(16))) _Float16 v16h;
typedef __attribute__((ext_vector_type(8)))  _Float16 v8h;
typedef __attribute__((ext_vector_type(4)))  _Float16 v4h;
typedef __attribute__((ext_vector_type(8)))  float    v8f;

#define D_MODEL 1024
#define NH      16
#define DK      64
#define BB      4
#define SS      2048
#define MROWS   (BB * SS)     // 8192 rows of activations

// ---------------------------------------------------------------------------
// Fragment loaders (CDNA5 wave32 WMMA layouts, CDNA5 ISA §7.12.2)
// A 16x32 f16: lane row = lane&15; elems = [k0..k0+7] and [k0+16..k0+23],
//              k0 = (lane>>4)*8  -> two 16B chunks.
// B 32x16 f16: lane col = lane&15; elems = [k0..k0+7] and [k0+16..k0+23],
//              k0 = (lane>>4)*8  -> two 16B chunks (same as A).
// ---------------------------------------------------------------------------
__device__ __forceinline__ v16h load_afrag(const _Float16* p) {
  v8h lo = *(const v8h*)(p);
  v8h hi = *(const v8h*)(p + 16);
  return __builtin_shufflevector(lo, hi, 0,1,2,3,4,5,6,7,8,9,10,11,12,13,14,15);
}
__device__ __forceinline__ v16h load_bfrag(const _Float16* p) {
  v8h lo = *(const v8h*)(p);
  v8h hi = *(const v8h*)(p + 16);
  return __builtin_shufflevector(lo, hi, 0,1,2,3,4,5,6,7,8,9,10,11,12,13,14,15);
}
__device__ __forceinline__ v8f wmma_f16(v16h a, v16h b, v8f c) {
  return __builtin_amdgcn_wmma_f32_16x16x32_f16(false, a, false, b,
                                                (short)0, c, false, false);
}

// ---------------------------------------------------------------------------
// fp32 -> f16 streaming cast (n multiple of 4)
// ---------------------------------------------------------------------------
__global__ __launch_bounds__(256)
void cast_f32_to_f16(const float* __restrict__ in, _Float16* __restrict__ out, int n) {
  int i = (blockIdx.x * blockDim.x + threadIdx.x) * 4;
  if (i < n) {
    float4 v = *(const float4*)(in + i);
    v4h o = { (_Float16)v.x, (_Float16)v.y, (_Float16)v.z, (_Float16)v.w };
    *(v4h*)(out + i) = o;
  }
}

// ---------------------------------------------------------------------------
// C[m][n] = A[m][:] . W[n][:] + bias[n]   (torch Linear: x @ W.T + b)
// A: MROWS x 1024 f16 row-major, W: 1024 x 1024 f16 row-major.
// Block = 8 waves (4x2) -> 128x128 tile; wave = 32x64 tile (2x4 WMMA accs).
// Per k-step: 4 A-loads + 8 B-loads (b128) feed 8 WMMAs (1.5 loads/WMMA).
// mode 0: write f16 to [b,h,s,d]          (q / k for attention)
// mode 1: write f16 to [b,h,d,s]          (v transposed for P.V fragments)
// mode 2: write f32 row-major to d_out    (final projection)
// ---------------------------------------------------------------------------
__device__ __forceinline__ void store_tile(v8f c, int m0, int n0,
                                           const float* __restrict__ bias,
                                           void* __restrict__ out, int mode, int lane) {
  const int mrow = (lane >> 4) << 3;
  const int ncol = lane & 15;
  const int n = n0 + ncol;
  const float bn = bias[n];
  #pragma unroll
  for (int r = 0; r < 8; ++r) {
    const int m = m0 + mrow + r;
    const float val = c[r] + bn;
    if (mode == 2) {
      ((float*)out)[(size_t)m * D_MODEL + n] = val;
    } else {
      const int b = m >> 11, s = m & (SS - 1);
      const int h = n >> 6,  d = n & (DK - 1);
      _Float16* o = (_Float16*)out;
      if (mode == 0)
        o[(((size_t)(b * NH + h) * SS + s) * DK) + d] = (_Float16)val;
      else
        o[(((size_t)(b * NH + h) * DK + d) * SS) + s] = (_Float16)val;
    }
  }
}

__global__ __launch_bounds__(256)
void gemm_wmma(const _Float16* __restrict__ A, const _Float16* __restrict__ W,
               const float* __restrict__ bias, void* __restrict__ out, int mode) {
  const int lane = threadIdx.x & 31;
  const int wave = threadIdx.x >> 5;
  const int wm = wave >> 1, wn = wave & 1;
  const int m0 = blockIdx.x * 128 + wm * 32;
  const int n0 = blockIdx.y * 128 + wn * 64;
  const int rsel = lane & 15;
  const int hsel = lane >> 4;

  const _Float16* Ab0 = A + (size_t)(m0 + rsel) * D_MODEL + hsel * 8;
  const _Float16* Ab1 = Ab0 + (size_t)16 * D_MODEL;
  const _Float16* Wb  = W + (size_t)(n0 + rsel) * D_MODEL + hsel * 8;

  v8f c0[4] = {{}, {}, {}, {}};
  v8f c1[4] = {{}, {}, {}, {}};

  #pragma unroll 2
  for (int k = 0; k < D_MODEL; k += 32) {
    v16h a0 = load_afrag(Ab0 + k);
    v16h a1 = load_afrag(Ab1 + k);
    #pragma unroll
    for (int t = 0; t < 4; ++t) {
      v16h b = load_bfrag(Wb + (size_t)(t * 16) * D_MODEL + k);
      c0[t] = wmma_f16(a0, b, c0[t]);
      c1[t] = wmma_f16(a1, b, c1[t]);
    }
  }
  #pragma unroll
  for (int t = 0; t < 4; ++t) {
    store_tile(c0[t], m0,      n0 + 16 * t, bias, out, mode, lane);
    store_tile(c1[t], m0 + 16, n0 + 16 * t, bias, out, mode, lane);
  }
}

// ---------------------------------------------------------------------------
// Causal flash attention. One wave per 16-query tile per (batch,head).
// qh/kh: [b,h,s,d] f16; vth: [b,h,d,s] f16; ah out: [(b*S+s), h*64+d] f16.
// Key step = 64: 8 WMMAs for scores (16x64), online softmax with cross-lane
// shfl_xor row reductions amortized over 64 keys, P relayout via per-wave
// LDS (in-order DS within wave32), 8 WMMAs for P.V.
// 1/sqrt(DK) is pre-folded into the Q fragments (softmax is invariant).
// ---------------------------------------------------------------------------
#define PSTR 72   // LDS row stride in halfs (16B-aligned chunks, conflict pad)

__global__ __launch_bounds__(32)
void attn_flash(const _Float16* __restrict__ qh, const _Float16* __restrict__ kh,
                const _Float16* __restrict__ vth, _Float16* __restrict__ ah) {
  __shared__ _Float16 pls[16 * PSTR];

  const int lane = threadIdx.x & 31;
  const int qt = blockIdx.x & 127;          // S/16 query tiles
  const int bh = blockIdx.x >> 7;           // 0..63
  const int q0 = qt << 4;
  const int rsel = lane & 15;
  const int hsel = lane >> 4;

  const _Float16* Q  = qh  + (size_t)bh * SS * DK;
  const _Float16* K  = kh  + (size_t)bh * SS * DK;
  const _Float16* VT = vth + (size_t)bh * DK * SS;

  // Q fragments (d = 0..31, 32..63), pre-scaled by 1/sqrt(64)
  const _Float16* qrow = Q + (size_t)(q0 + rsel) * DK + hsel * 8;
  v16h aq0 = load_afrag(qrow)      * (_Float16)0.125f;
  v16h aq1 = load_afrag(qrow + 32) * (_Float16)0.125f;

  v8f acc0 = {}, acc1 = {}, acc2 = {}, acc3 = {};
  float mrow[8], lrow[8];
  #pragma unroll
  for (int r = 0; r < 8; ++r) { mrow[r] = -1e30f; lrow[r] = 0.0f; }

  for (int kk = 0; kk <= q0; kk += 64) {
    // ---- scores: 16 queries x 64 keys, four 16x16 C tiles ----------------
    v8f s[4] = {{}, {}, {}, {}};
    #pragma unroll
    for (int j = 0; j < 4; ++j) {
      const _Float16* kr = K + (size_t)(kk + 16 * j + rsel) * DK + hsel * 8;
      s[j] = wmma_f16(aq0, load_bfrag(kr),      s[j]);
      s[j] = wmma_f16(aq1, load_bfrag(kr + 32), s[j]);
    }

    // ---- causal mask (only the block containing the diagonal) ------------
    const bool tail = (kk + 64 > q0);
    if (tail) {
      #pragma unroll
      for (int j = 0; j < 4; ++j) {
        #pragma unroll
        for (int r = 0; r < 8; ++r) {
          const int qi = q0 + (hsel << 3) + r;
          if (kk + 16 * j + rsel > qi) s[j][r] = -1e30f;
        }
      }
    }

    // ---- online softmax (rows striped: row = r + 8*hsel, cols on lanes) --
    #pragma unroll
    for (int r = 0; r < 8; ++r) {
      float t = fmaxf(fmaxf(s[0][r], s[1][r]), fmaxf(s[2][r], s[3][r]));
      t = fmaxf(t, __shfl_xor(t, 1));
      t = fmaxf(t, __shfl_xor(t, 2));
      t = fmaxf(t, __shfl_xor(t, 4));
      t = fmaxf(t, __shfl_xor(t, 8));
      const float mn    = fmaxf(mrow[r], t);
      const float alpha = __expf(mrow[r] - mn);
      mrow[r] = mn;
      const float p0 = __expf(s[0][r] - mn);
      const float p1 = __expf(s[1][r] - mn);
      const float p2 = __expf(s[2][r] - mn);
      const float p3 = __expf(s[3][r] - mn);
      float u = (p0 + p1) + (p2 + p3);
      u += __shfl_xor(u, 1);
      u += __shfl_xor(u, 2);
      u += __shfl_xor(u, 4);
      u += __shfl_xor(u, 8);
      lrow[r] = lrow[r] * alpha + u;
      acc0[r] *= alpha; acc1[r] *= alpha; acc2[r] *= alpha; acc3[r] *= alpha;
      const int prow = (hsel << 3) + r;
      pls[prow * PSTR +      rsel] = (_Float16)p0;
      pls[prow * PSTR + 16 + rsel] = (_Float16)p1;
      pls[prow * PSTR + 32 + rsel] = (_Float16)p2;
      pls[prow * PSTR + 48 + rsel] = (_Float16)p3;
    }
    asm volatile("" ::: "memory");   // keep DS store -> DS load ordered in IR

    // ---- P as two A-fragments (16 queries x 64 keys) ---------------------
    v16h ap0, ap1;
    {
      const _Float16* pp = &pls[rsel * PSTR + hsel * 8];
      v8h lo0 = *(const v8h*)(pp);
      v8h hi0 = *(const v8h*)(pp + 16);
      v8h lo1 = *(const v8h*)(pp + 32);
      v8h hi1 = *(const v8h*)(pp + 48);
      ap0 = __builtin_shufflevector(lo0, hi0, 0,1,2,3,4,5,6,7,8,9,10,11,12,13,14,15);
      ap1 = __builtin_shufflevector(lo1, hi1, 0,1,2,3,4,5,6,7,8,9,10,11,12,13,14,15);
    }

    // ---- P . V : B fragments contiguous along s in VT --------------------
    const _Float16* vb = VT + kk + hsel * 8;
    #pragma unroll
    for (int t = 0; t < 4; ++t) {
      const _Float16* vr = vb + (size_t)(16 * t + rsel) * SS;
      v8f* a = (t == 0) ? &acc0 : (t == 1) ? &acc1 : (t == 2) ? &acc2 : &acc3;
      *a = wmma_f16(ap0, load_bfrag(vr),      *a);
      *a = wmma_f16(ap1, load_bfrag(vr + 32), *a);
    }
  }

  // ---- normalize and store to [(b*S+s), h*64+d] f16 ----------------------
  const int b = bh >> 4, h = bh & 15;
  #pragma unroll
  for (int r = 0; r < 8; ++r) {
    const int srow = q0 + (hsel << 3) + r;
    const float inv = 1.0f / lrow[r];
    _Float16* orow = ah + ((size_t)b * SS + srow) * D_MODEL + h * DK;
    orow[rsel]      = (_Float16)(acc0[r] * inv);
    orow[16 + rsel] = (_Float16)(acc1[r] * inv);
    orow[32 + rsel] = (_Float16)(acc2[r] * inv);
    orow[48 + rsel] = (_Float16)(acc3[r] * inv);
  }
}

// ---------------------------------------------------------------------------
// Host-side orchestration
// ---------------------------------------------------------------------------
extern "C" void kernel_launch(void* const* d_in, const int* in_sizes, int n_in,
                              void* d_out, int out_size, void* d_ws, size_t ws_size,
                              hipStream_t stream) {
  const float* Q  = (const float*)d_in[0];
  const float* K  = (const float*)d_in[1];
  const float* V  = (const float*)d_in[2];
  // d_in[3] = mask (causal, handled analytically in-kernel)
  const float* Wq = (const float*)d_in[4];
  const float* bq = (const float*)d_in[5];
  const float* Wk = (const float*)d_in[6];
  const float* bk = (const float*)d_in[7];
  const float* Wv = (const float*)d_in[8];
  const float* bv = (const float*)d_in[9];
  const float* Wo = (const float*)d_in[10];
  const float* bo = (const float*)d_in[11];

  const size_t NACT = (size_t)MROWS * D_MODEL;     // 8 Mi elems
  const size_t NW   = (size_t)D_MODEL * D_MODEL;   // 1 Mi elems

  _Float16* ws  = (_Float16*)d_ws;
  _Float16* Qh  = ws;
  _Float16* Kh  = Qh  + NACT;
  _Float16* Vh  = Kh  + NACT;
  _Float16* Wqh = Vh  + NACT;
  _Float16* Wkh = Wqh + NW;
  _Float16* Wvh = Wkh + NW;
  _Float16* Woh = Wvh + NW;
  _Float16* qh  = Woh + NW;     // [b,h,s,d]
  _Float16* kh  = qh  + NACT;   // [b,h,s,d]
  _Float16* vth = kh  + NACT;   // [b,h,d,s]
  _Float16* ah  = vth + NACT;   // [(b*S+s), 1024]

  auto cast = [&](const float* src, _Float16* dst, size_t n) {
    unsigned blocks = (unsigned)((n / 4 + 255) / 256);
    cast_f32_to_f16<<<dim3(blocks), dim3(256), 0, stream>>>(src, dst, (int)n);
  };
  cast(Q,  Qh,  NACT);
  cast(K,  Kh,  NACT);
  cast(V,  Vh,  NACT);
  cast(Wq, Wqh, NW);
  cast(Wk, Wkh, NW);
  cast(Wv, Wvh, NW);
  cast(Wo, Woh, NW);

  dim3 gg(MROWS / 128, D_MODEL / 128), gb(256);
  gemm_wmma<<<gg, gb, 0, stream>>>(Qh, Wqh, bq, qh,  0);
  gemm_wmma<<<gg, gb, 0, stream>>>(Kh, Wkh, bk, kh,  0);
  gemm_wmma<<<gg, gb, 0, stream>>>(Vh, Wvh, bv, vth, 1);

  attn_flash<<<dim3(BB * NH * (SS / 16)), dim3(32), 0, stream>>>(qh, kh, vth, ah);

  gemm_wmma<<<gg, gb, 0, stream>>>(ah, Woh, bo, d_out, 2);
}
